// Frequency_Enhanced_Block_59373627900513
// MI455X (gfx1250) — matrix operand order repaired and checked by reading, not compiled, not run on hardware
//
#include <hip/hip_runtime.h>
#include <hip/hip_bf16.h>

// ---------------------------------------------------------------------------
// Frequency-Enhanced Block (FEDformer) for MI455X / gfx1250, wave32 + WMMA.
// B=32, L=2048, D=512, MODES=32, KMA=128.
// ---------------------------------------------------------------------------

typedef __bf16 bf16_t;
typedef __bf16 v16bf __attribute__((ext_vector_type(16)));
typedef __bf16 v8bf  __attribute__((ext_vector_type(8)));
typedef float  v8f   __attribute__((ext_vector_type(8)));
typedef float  v4f   __attribute__((ext_vector_type(4)));

#define LDSPAD   8                 // bf16 pad per 32-wide row -> 80B row stride
#define RSTRIDE  (32 + LDSPAD)     // 40 bf16 elements

// ---------------------------------------------------------------------------
// Generic bf16 WMMA GEMM:  C[z] = alpha * A[z] * B[z]  (+ bias) (+ resid) (gelu)
// A: row-major M x K (lda).  B element (k,n) at B + k*bStrideK + n*bStrideN.
//   BMODE 0: bStrideK==1  (weight W[N][K] row-major)   -> float4 loads over k
//   BMODE 1: bStrideN==1  (math-B [K][N] row-major)    -> float4 loads over n
//   BMODE 2: generic strides (mode-innermost Wf, L2-resident)
// Tile: (32*WM) x (64*WN), 8 waves, each wave 2x4 WMMA 16x16 subtiles.
// M, N, K must be multiples of the tile sizes (true for every launch below).
// Epilogue shape (bias / residual / gelu) is compile-time for clean codegen.
// ---------------------------------------------------------------------------
template<int WM, int WN, int BMODE, bool HAS_BIAS, bool HAS_RES, bool ACT>
__global__ __launch_bounds__(256) void gemm_bf16_wmma(
    const float* __restrict__ A, int lda, long sAz,
    const float* __restrict__ B, long bStrideK, long bStrideN, long sBz,
    float* __restrict__ C, int ldc, long sCz,
    const float* __restrict__ R, int ldr, long sRz,
    const float* __restrict__ bias,
    int K, float alpha)
{
    constexpr int TM = 32 * WM;
    constexpr int TN = 64 * WN;
    __shared__ bf16_t Asl[TM * RSTRIDE];
    __shared__ bf16_t Bsl[TN * RSTRIDE];

    const int tid  = threadIdx.x;
    const int lane = tid & 31;
    const int wave = tid >> 5;
    const int wm   = wave % WM;
    const int wn   = wave / WM;
    const int ll   = lane & 15;     // N / M position within subtile
    const int lh   = lane >> 4;     // K-half selector (A/B frag layouts)

    const int nBase = blockIdx.x * TN;
    const int mBase = blockIdx.y * TM;
    const int z     = blockIdx.z;

    const float* Ab = A + (size_t)z * sAz + (size_t)mBase * lda;
    const float* Bb = B + (size_t)z * sBz;
    float*       Cb = C + (size_t)z * sCz;
    const float* Rb = HAS_RES ? (R + (size_t)z * sRz) : nullptr;

    v8f acc[2][4];
    #pragma unroll
    for (int i = 0; i < 2; ++i)
        #pragma unroll
        for (int j = 0; j < 4; ++j)
            #pragma unroll
            for (int r = 0; r < 8; ++r) acc[i][j][r] = 0.0f;

    for (int kb = 0; kb < K; kb += 32) {
        const bool pf = (kb + 32 < K);
        // ---- stage A tile: TM x 32 floats -> bf16 LDS [m][k] -------------
        {
            constexpr int NV4 = TM * 8;          // float4 count
            #pragma unroll
            for (int e = 0; e < NV4 / 256; ++e) {
                int i4 = tid + e * 256;
                int r  = i4 >> 3;
                int c4 = i4 & 7;
                const float* gp = Ab + (size_t)r * lda + kb + c4 * 4;
                if (pf) __builtin_prefetch(gp + 32, 0, 1);
                v4f v = *(const v4f*)gp;
                bf16_t* d = &Asl[r * RSTRIDE + c4 * 4];
                d[0] = (bf16_t)v[0]; d[1] = (bf16_t)v[1];
                d[2] = (bf16_t)v[2]; d[3] = (bf16_t)v[3];
            }
        }
        // ---- stage B tile -> bf16 LDS Bt[n][k] ---------------------------
        if (BMODE == 0) {
            constexpr int NV4 = TN * 8;
            #pragma unroll 2
            for (int e = 0; e < NV4 / 256; ++e) {
                int i4 = tid + e * 256;
                int n  = i4 >> 3;
                int c4 = i4 & 7;
                const float* gp = Bb + (size_t)(nBase + n) * bStrideN + kb + c4 * 4;
                if (pf) __builtin_prefetch(gp + 32, 0, 1);
                v4f v = *(const v4f*)gp;
                bf16_t* d = &Bsl[n * RSTRIDE + c4 * 4];
                d[0] = (bf16_t)v[0]; d[1] = (bf16_t)v[1];
                d[2] = (bf16_t)v[2]; d[3] = (bf16_t)v[3];
            }
        } else if (BMODE == 1) {
            constexpr int NV4 = TN * 8;          // 32 k-rows * TN/4 float4s
            #pragma unroll 2
            for (int e = 0; e < NV4 / 256; ++e) {
                int i4 = tid + e * 256;
                int k  = i4 / (TN / 4);
                int n4 = i4 % (TN / 4);
                v4f v = *(const v4f*)(Bb + (size_t)(kb + k) * bStrideK + nBase + n4 * 4);
                #pragma unroll
                for (int u = 0; u < 4; ++u)
                    Bsl[(n4 * 4 + u) * RSTRIDE + k] = (bf16_t)v[u];
            }
        } else {
            constexpr int NE = TN * 32;
            for (int e = 0; e < NE / 256; ++e) {
                int i = tid + e * 256;
                int n = i % TN;
                int k = i / TN;
                Bsl[n * RSTRIDE + k] =
                    (bf16_t)Bb[(size_t)(kb + k) * bStrideK + (size_t)(nBase + n) * bStrideN];
            }
        }
        __syncthreads();

        // ---- fragments per documented CDNA5 16-bit layouts ---------------
        // A 16x32: lane ll = M row; lh half: elems 0..7 = K 8*lh..+7,
        //          elems 8..15 = K 16+8*lh..+7.
// B 32x16: lane ll = N col; elems 0..7 = K 8*lh..+7, elems 8..15 = K 16+8*lh..+7.
        v16bf aF[2], bF[4];
        #pragma unroll
        for (int sm = 0; sm < 2; ++sm) {
            const int m = 32 * wm + 16 * sm + ll;
            v8bf lo = *(const v8bf*)&Asl[m * RSTRIDE + 8 * lh];
            v8bf hi = *(const v8bf*)&Asl[m * RSTRIDE + 16 + 8 * lh];
            aF[sm] = __builtin_shufflevector(lo, hi,
                     0,1,2,3,4,5,6,7,8,9,10,11,12,13,14,15);
        }
        #pragma unroll
        for (int sn = 0; sn < 4; ++sn) {
            const int n = 64 * wn + 16 * sn + ll;
            v8bf lo = *(const v8bf*)&Bsl[n * RSTRIDE + 8 * lh];
            v8bf hi = *(const v8bf*)&Bsl[n * RSTRIDE + 16 + 8 * lh];
            bF[sn] = __builtin_shufflevector(lo, hi,
                     0,1,2,3,4,5,6,7,8,9,10,11,12,13,14,15);
        }
        #pragma unroll
        for (int sm = 0; sm < 2; ++sm)
            #pragma unroll
            for (int sn = 0; sn < 4; ++sn)
                acc[sm][sn] = __builtin_amdgcn_wmma_f32_16x16x32_bf16(
                    false, aF[sm], false, bF[sn], (short)0, acc[sm][sn],
                    false, false);
        __syncthreads();
    }

    // ---- epilogue: C layout lane ll = N, VGPR r -> M = 8*lh + r ----------
    #pragma unroll
    for (int sm = 0; sm < 2; ++sm) {
        #pragma unroll
        for (int sn = 0; sn < 4; ++sn) {
            const int gn  = nBase + 64 * wn + 16 * sn + ll;
            const int gm0 = mBase + 32 * wm + 16 * sm + 8 * lh;
            const float bv = HAS_BIAS ? bias[gn] : 0.0f;
            float*       cp = Cb + (size_t)gm0 * ldc + gn;
            const float* rp = HAS_RES ? (Rb + (size_t)gm0 * ldr + gn) : nullptr;
            #pragma unroll
            for (int r = 0; r < 8; ++r) {
                float v = alpha * acc[sm][sn][r] + bv;
                if (HAS_RES) v += rp[(size_t)r * ldr];
                if (ACT)     v = 0.5f * v * (1.0f + erff(v * 0.70710678118654752f));
                cp[(size_t)r * ldc] = v;
            }
        }
    }
}

// ---------------------------------------------------------------------------
// DFT basis Fb[64][2048]: rows 0..31 = cos(w m t), rows 32..63 = -sin(w m t)
// Exact angle reduction via (m*t mod 2048).
// ---------------------------------------------------------------------------
__global__ __launch_bounds__(256) void fill_fb_kernel(float* __restrict__ Fb)
{
    int idx = blockIdx.x * 256 + threadIdx.x;     // 64*2048 exact
    int m = idx >> 11;
    int t = idx & 2047;
    int mm = (m < 32) ? m : m - 32;
    int p = (mm * t) & 2047;
    float ang = (float)p * (float)(2.0 * 3.14159265358979323846 / 2048.0);
    float s, c;
    sincosf(ang, &s, &c);
    Fb[idx] = (m < 32) ? c : -s;
}

// iDFT basis Gt[2048][64]: k<32 -> ((k==0)?1:2)/L * cos(w k t)
//                          k>=32, m=k-32 -> (m==0)?0:-2/L * sin(w m t)
__global__ __launch_bounds__(256) void fill_gt_kernel(float* __restrict__ Gt)
{
    int idx = blockIdx.x * 256 + threadIdx.x;     // 2048*64 exact
    int t = idx >> 6;
    int k = idx & 63;
    int m = (k < 32) ? k : k - 32;
    int p = (m * t) & 2047;
    float ang = (float)p * (float)(2.0 * 3.14159265358979323846 / 2048.0);
    float s, c;
    sincosf(ang, &s, &c);
    const float inv = 1.0f / 2048.0f;
    float val;
    if (k < 32)       val = ((k == 0) ? 1.0f : 2.0f) * inv * c;
    else              val = (m == 0) ? 0.0f : -2.0f * inv * s;
    Gt[idx] = val;
}

// ---------------------------------------------------------------------------
// Series decomposition residual: out = in - movavg_128(in), replicate-padded.
// window for index t is j in [t-64, t+63] with clamp to [0, L-1].
// grid: (tChunks=8, dChunks=2, b=32); 256 threads over d.
// ---------------------------------------------------------------------------
__global__ __launch_bounds__(256) void ma_sub_kernel(
    const float* __restrict__ in, float* __restrict__ out)
{
    const int tid = threadIdx.x;
    const int d   = blockIdx.y * 256 + tid;
    const int b   = blockIdx.z;
    const int t0  = blockIdx.x * 256;
    const float* base  = in  + (size_t)b * 2048 * 512 + d;
    float*       obase = out + (size_t)b * 2048 * 512 + d;
    auto at = [&](int t) {
        t = t < 0 ? 0 : (t > 2047 ? 2047 : t);
        return base[(size_t)t * 512];
    };
    float s = 0.0f;
    for (int j = t0 - 64; j <= t0 + 63; ++j) s += at(j);
    for (int i = 0; i < 256; ++i) {
        const int t = t0 + i;
        const float center = base[(size_t)t * 512];
        obase[(size_t)t * 512] = center - s * (1.0f / 128.0f);
        s += at(t + 64) - at(t - 64);
    }
}

// ---------------------------------------------------------------------------
// BatchNorm over time axis: stats per l over (b, d) = 16384 values.
// ---------------------------------------------------------------------------
__global__ __launch_bounds__(256) void bn_stats_kernel(
    const float* __restrict__ res, float* __restrict__ mean,
    float* __restrict__ rstd)
{
    __shared__ float sh_s[256];
    __shared__ float sh_q[256];
    const int l = blockIdx.x;
    const int tid = threadIdx.x;
    float s = 0.0f, q = 0.0f;
    for (int b = 0; b < 32; ++b) {
        const float* p = res + ((size_t)b * 2048 + l) * 512;
        for (int d = tid; d < 512; d += 256) {
            float v = p[d];
            s += v; q += v * v;
        }
    }
    sh_s[tid] = s; sh_q[tid] = q;
    __syncthreads();
    for (int off = 128; off > 0; off >>= 1) {
        if (tid < off) { sh_s[tid] += sh_s[tid + off]; sh_q[tid] += sh_q[tid + off]; }
        __syncthreads();
    }
    if (tid == 0) {
        float mu  = sh_s[0] * (1.0f / 16384.0f);
        float var = sh_q[0] * (1.0f / 16384.0f) - mu * mu;
        mean[l] = mu;
        rstd[l] = rsqrtf(var + 1e-5f);
    }
}

__global__ __launch_bounds__(256) void bn_apply_kernel(
    float* __restrict__ res, const float* __restrict__ gamma,
    const float* __restrict__ beta, const float* __restrict__ mean,
    const float* __restrict__ rstd)
{
    size_t i4  = (size_t)blockIdx.x * 256 + threadIdx.x;
    size_t off = i4 * 4;
    int l = (int)((off >> 9) & 2047);
    float g = gamma[l] * rstd[l];
    float bb = beta[l] - mean[l] * g;
    v4f v = *(const v4f*)(res + off);
    v4f o;
    #pragma unroll
    for (int u = 0; u < 4; ++u) o[u] = v[u] * g + bb;
    *(v4f*)(res + off) = o;
}

// ---------------------------------------------------------------------------
extern "C" void kernel_launch(void* const* d_in, const int* in_sizes, int n_in,
                              void* d_out, int out_size, void* d_ws, size_t ws_size,
                              hipStream_t stream)
{
    (void)in_sizes; (void)n_in; (void)out_size; (void)ws_size;
    const float* x      = (const float*)d_in[0];   // (32, 2048, 512)
    const float* W_in   = (const float*)d_in[1];   // (512, 512) [o][d]
    const float* b_in   = (const float*)d_in[2];   // (512)
    const float* Wf_r   = (const float*)d_in[3];   // (1, 512, 512, 32) [i][o][m]
    const float* Wf_i   = (const float*)d_in[4];
    const float* W_out  = (const float*)d_in[5];   // (512, 512)
    const float* b_out  = (const float*)d_in[6];
    const float* W_c1   = (const float*)d_in[7];   // (512, 512)
    const float* W_c2   = (const float*)d_in[8];
    const float* gamma  = (const float*)d_in[9];   // (2048)
    const float* beta   = (const float*)d_in[10];
    float* out = (float*)d_out;                    // (32, 2048, 512)

    const long BLD = 32L * 2048 * 512;             // 33,554,432
    float* w    = (float*)d_ws;
    float* buf0 = w;                               // BLD floats
    float* Xf   = buf0 + BLD;                      // 32*64*512
    float* OM   = Xf + 32L * 64 * 512;             // 32*64*512
    float* Fb   = OM + 32L * 64 * 512;             // 64*2048
    float* Gt   = Fb + 64L * 2048;                 // 2048*64
    float* bnM  = Gt + 2048L * 64;                 // 2048
    float* bnR  = bnM + 2048;                      // 2048

    // 0) DFT / iDFT bases
    fill_fb_kernel<<<512, 256, 0, stream>>>(Fb);
    fill_gt_kernel<<<512, 256, 0, stream>>>(Gt);

    // 1) q = x @ W_in^T + b_in  -> buf0   (65536 x 512 x 512)
    gemm_bf16_wmma<4, 2, 0, true, false, false>
        <<<dim3(4, 512, 1), 256, 0, stream>>>(
        x, 512, 0L, W_in, 1L, 512L, 0L,
        buf0, 512, 0L, nullptr, 0, 0L, b_in, 512, 1.0f);

    // 2) Truncated rfft: Xf[b] (64x512) = Fb (64x2048) * q_b (2048x512)
    gemm_bf16_wmma<2, 4, 1, false, false, false>
        <<<dim3(2, 1, 32), 256, 0, stream>>>(
        Fb, 2048, 0L, buf0, 512L, 1L, (long)2048 * 512,
        Xf, 512, 64L * 512, nullptr, 0, 0L, nullptr, 2048, 1.0f);

    // 3) Mode mixing (complex), 32 GEMMs of M=32(b), N=512(o), K=512(i).
    //    Xr at Xf+0, Xi at Xf+32*512 (row stride over b = 64*512).
    //    OMr = Xr*Wr - Xi*Wi ; OMi = Xr*Wi + Xi*Wr.
    const long XLDA = 64L * 512;      // b-row stride
    gemm_bf16_wmma<1, 8, 2, false, false, false>
        <<<dim3(1, 1, 32), 256, 0, stream>>>(
        Xf, (int)XLDA, 512L, Wf_r, 512L * 32, 32L, 1L,
        OM, (int)XLDA, 512L, nullptr, 0, 0L, nullptr, 512, 1.0f);
    gemm_bf16_wmma<1, 8, 2, false, true, false>
        <<<dim3(1, 1, 32), 256, 0, stream>>>(
        Xf + 32L * 512, (int)XLDA, 512L, Wf_i, 512L * 32, 32L, 1L,
        OM, (int)XLDA, 512L, OM, (int)XLDA, 512L, nullptr, 512, -1.0f);
    gemm_bf16_wmma<1, 8, 2, false, false, false>
        <<<dim3(1, 1, 32), 256, 0, stream>>>(
        Xf, (int)XLDA, 512L, Wf_i, 512L * 32, 32L, 1L,
        OM + 32L * 512, (int)XLDA, 512L, nullptr, 0, 0L, nullptr, 512, 1.0f);
    gemm_bf16_wmma<1, 8, 2, false, true, false>
        <<<dim3(1, 1, 32), 256, 0, stream>>>(
        Xf + 32L * 512, (int)XLDA, 512L, Wf_r, 512L * 32, 32L, 1L,
        OM + 32L * 512, (int)XLDA, 512L, OM + 32L * 512, (int)XLDA, 512L,
        nullptr, 512, 1.0f);

    // 4) irfft (32 modes): fre_b (2048x512) = Gt (2048x64) * OM_b (64x512) -> out
    gemm_bf16_wmma<4, 2, 1, false, false, false>
        <<<dim3(4, 16, 32), 256, 0, stream>>>(
        Gt, 64, 0L, OM, 512L, 1L, 64L * 512,
        out, 512, 2048L * 512, nullptr, 0, 0L, nullptr, 64, 1.0f);

    // 5) xr_pre = x + fre @ W_out^T + b_out -> buf0
    gemm_bf16_wmma<4, 2, 0, true, true, false>
        <<<dim3(4, 512, 1), 256, 0, stream>>>(
        out, 512, 0L, W_out, 1L, 512L, 0L,
        buf0, 512, 0L, x, 512, 0L, b_out, 512, 1.0f);

    // 6) xr = xr_pre - moving_average(xr_pre, 128) -> out
    ma_sub_kernel<<<dim3(8, 2, 32), 256, 0, stream>>>(buf0, out);

    // 7) y1 = gelu(xr @ W_conv1^T) -> buf0
    gemm_bf16_wmma<4, 2, 0, false, false, true>
        <<<dim3(4, 512, 1), 256, 0, stream>>>(
        out, 512, 0L, W_c1, 1L, 512L, 0L,
        buf0, 512, 0L, nullptr, 0, 0L, nullptr, 512, 1.0f);

    // 8) res = xr + y1 @ W_conv2^T -> out (in-place residual is per-element safe)
    gemm_bf16_wmma<4, 2, 0, false, true, false>
        <<<dim3(4, 512, 1), 256, 0, stream>>>(
        buf0, 512, 0L, W_c2, 1L, 512L, 0L,
        out, 512, 0L, out, 512, 0L, nullptr, 512, 1.0f);

    // 9) BatchNorm1d over time axis (stats over b,d per l), then normalize.
    bn_stats_kernel<<<2048, 256, 0, stream>>>(out, bnM, bnR);
    bn_apply_kernel<<<32768, 256, 0, stream>>>(out, gamma, beta, bnM, bnR);
}
